// GraphConv_2886218023535
// MI455X (gfx1250) — matrix operand order repaired and checked by reading, not compiled, not run on hardware
//
#include <hip/hip_runtime.h>

typedef __bf16  v4bf  __attribute__((ext_vector_type(4)));
typedef __bf16  v8bf  __attribute__((ext_vector_type(8)));
typedef __bf16  v16bf __attribute__((ext_vector_type(16)));
typedef float   v4f   __attribute__((ext_vector_type(4)));
typedef float   v8f   __attribute__((ext_vector_type(8)));

#define N_NODES   8192
#define C_IN      512
#define C_OUT     512
#define NEG_SLOPE 0.2f

// Assemble a 16x32 bf16 A-fragment (contiguous k order, k = 16*(L/16) + s):
// lane L: m = L%16; lanes 0-15 carry K 0-7 (v0-3) and 8-15 (v4-7),
// lanes 16-31 carry K 16-23 and 24-31.
static __device__ __forceinline__ v16bf make_afrag(const __bf16* lo, const __bf16* hi) {
    v8bf l = *(const v8bf*)lo;
    v8bf h = *(const v8bf*)hi;
    return __builtin_shufflevector(l, h, 0,1,2,3,4,5,6,7,8,9,10,11,12,13,14,15);
}

// ---------------------------------------------------------------------------
// Kernel 1: supportT[c][n] = bf16( (inputs @ w)[n][c] + b[c] )
// ---------------------------------------------------------------------------
__global__ __launch_bounds__(256) void support_kernel(
    const float* __restrict__ inputs, const float* __restrict__ w,
    const float* __restrict__ bias, unsigned short* __restrict__ supportT_raw)
{
    __bf16* __restrict__ supportT = (__bf16*)supportT_raw;
    __shared__ __align__(64) __bf16 lds_wT[C_OUT][32]; // w tile, transposed [c][k]
    __shared__ __align__(64) __bf16 lds_in[16][32];    // inputs tile [m][k]

    const int t    = threadIdx.x;
    const int lane = t & 31;
    const int wave = t >> 5;
    const int m0   = blockIdx.x * 16;
    const int cb   = wave * 64;
    const int lm   = lane & 15;
    const int kh   = lane >> 4;

    v8f acc[4] = {};

    for (int k0 = 0; k0 < C_IN; k0 += 32) {
        __syncthreads();
        // Stage w[k0..k0+31][0..511] -> lds_wT (bf16, transposed).
        #pragma unroll
        for (int ch = 0; ch < 4; ++ch) {
            v4f wv[4];
            int r[4], c4[4];
            #pragma unroll
            for (int i = 0; i < 4; ++i) {
                int f  = t + ((ch * 4 + i) << 8);   // float4 index 0..4095
                r[i]   = f >> 7;                     // k row 0..31
                c4[i]  = (f & 127) << 2;             // col 0..508
                wv[i]  = *(const v4f*)&w[(size_t)(k0 + r[i]) * C_OUT + c4[i]];
            }
            #pragma unroll
            for (int i = 0; i < 4; ++i)
                #pragma unroll
                for (int j = 0; j < 4; ++j)
                    lds_wT[c4[i] + j][r[i]] = (__bf16)wv[i][j];
        }
        // Stage inputs[m0..m0+15][k0..k0+31]
        if (t < 128) {
            int r  = t >> 3;
            int c4 = (t & 7) << 2;
            v4f iv = *(const v4f*)&inputs[(size_t)(m0 + r) * C_IN + k0 + c4];
            v4bf p;
            #pragma unroll
            for (int j = 0; j < 4; ++j) p[j] = (__bf16)iv[j];
            *(v4bf*)&lds_in[r][c4] = p;
        }
        __syncthreads();

        // Batch ALL fragment loads, fence the scheduler, then WMMA burst.
        v16bf a = make_afrag(&lds_in[lm][kh * 16], &lds_in[lm][kh * 16 + 8]);
        v16bf bfr[4];
        #pragma unroll
        for (int nt = 0; nt < 4; ++nt)
            bfr[nt] = *(const v16bf*)&lds_wT[cb + nt * 16 + lm][kh * 16];
        __builtin_amdgcn_sched_barrier(0);   // loads stay above, WMMAs below
        #pragma unroll
        for (int nt = 0; nt < 4; ++nt)
            acc[nt] = __builtin_amdgcn_wmma_f32_16x16x32_bf16(
                false, a, false, bfr[nt], (short)0, acc[nt], false, false);
        __builtin_amdgcn_sched_barrier(0);
    }

    #pragma unroll
    for (int nt = 0; nt < 4; ++nt) {
        int col  = cb + nt * 16 + lm;
        float bv = bias[col];
        #pragma unroll
        for (int r = 0; r < 8; ++r) {
            int row = m0 + (kh ? 8 : 0) + r;
            supportT[(size_t)col * N_NODES + row] = (__bf16)(acc[nt][r] + bv);
        }
    }
}

// ---------------------------------------------------------------------------
// Kernel 2: out = LeakyReLU( (Σk att[k]*adj[k]) @ support )
// Pipelined: next adj tile loads into registers under the WMMAs; all
// fragment loads are batched ahead of the 16-WMMA burst, with scheduler
// fences so the backend cannot re-serialize load->wait->wmma.
// ---------------------------------------------------------------------------
__global__ __launch_bounds__(256) void graphconv_kernel(
    const float* __restrict__ adj, const float* __restrict__ att,
    const unsigned short* __restrict__ supportT_raw, float* __restrict__ out)
{
    const __bf16* __restrict__ supportT = (const __bf16*)supportT_raw;
    __shared__ __align__(64) __bf16 lds_a[32][64];  // combined-adj tile (bf16)

    const int t    = threadIdx.x;
    const int lane = t & 31;
    const int wave = t >> 5;
    const int m0   = blockIdx.x * 32;
    const int cb   = wave * 64;
    const int lm   = lane & 15;
    const int kh   = lane >> 4;

    const float a0 = att[0], a1 = att[1], a2 = att[2], a3 = att[3];
    const size_t NN = (size_t)N_NODES * N_NODES;

    const int cr = t >> 3;          // cooperative row 0..31
    const int cc = (t & 7) << 2;    // cooperative col base 0..28
    const size_t rowoff = (size_t)(m0 + cr) * N_NODES + (size_t)cc;

    v8f acc[2][4] = {};
    v4f pre[2][4];                  // prefetched adj tile: [half][matrix]

    // Prologue: issue loads for the first K-tile (k0 = 0).
    #pragma unroll
    for (int h2 = 0; h2 < 2; ++h2)
        #pragma unroll
        for (int mat = 0; mat < 4; ++mat)
            pre[h2][mat] = __builtin_nontemporal_load(
                (const v4f*)(adj + (size_t)mat * NN + rowoff + h2 * 32));

    for (int k0 = 0; k0 < N_NODES; k0 += 64) {
        __syncthreads();            // previous compute done reading lds_a
        // Combine prefetched tile with att weights, convert, store to LDS.
        #pragma unroll
        for (int h2 = 0; h2 < 2; ++h2) {
            v4f s = a0 * pre[h2][0] + a1 * pre[h2][1]
                  + a2 * pre[h2][2] + a3 * pre[h2][3];
            v4bf p;
            #pragma unroll
            for (int j = 0; j < 4; ++j) p[j] = (__bf16)s[j];
            *(v4bf*)&lds_a[cr][h2 * 32 + cc] = p;
        }
        __syncthreads();

        // Issue next tile's global loads NOW; they complete under the WMMAs.
        if (k0 + 64 < N_NODES) {
            size_t rb = rowoff + (size_t)(k0 + 64);
            #pragma unroll
            for (int h2 = 0; h2 < 2; ++h2)
                #pragma unroll
                for (int mat = 0; mat < 4; ++mat)
                    pre[h2][mat] = __builtin_nontemporal_load(
                        (const v4f*)(adj + (size_t)mat * NN + rb + h2 * 32));
        }
        __builtin_amdgcn_sched_barrier(0);   // keep adj prefetch issued early

        // Batch ALL fragment loads: 8 B-frags from L2-resident supportT
        // (16 global b128s) + 4 A-frags from LDS (8 ds_load_b128s) ...
        v16bf bfr[4][2];
        #pragma unroll
        for (int nt = 0; nt < 4; ++nt) {
            const __bf16* bp =
                &supportT[(size_t)(cb + nt * 16 + lm) * N_NODES + (size_t)(k0 + kh * 16)];
            bfr[nt][0] = *(const v16bf*)(bp);
            bfr[nt][1] = *(const v16bf*)(bp + 32);
        }
        v16bf afr[2][2];
        #pragma unroll
        for (int mt = 0; mt < 2; ++mt)
            #pragma unroll
            for (int ks = 0; ks < 2; ++ks)
                afr[mt][ks] = make_afrag(&lds_a[mt * 16 + lm][ks * 32 + kh * 16],
                                         &lds_a[mt * 16 + lm][ks * 32 + kh * 16 + 8]);
        __builtin_amdgcn_sched_barrier(0);   // loads above, WMMAs below

        // ... then a 16-WMMA burst (D->C accumulator chaining has no hazard).
        #pragma unroll
        for (int nt = 0; nt < 4; ++nt) {
            #pragma unroll
            for (int ks = 0; ks < 2; ++ks) {
                acc[0][nt] = __builtin_amdgcn_wmma_f32_16x16x32_bf16(
                    false, afr[0][ks], false, bfr[nt][ks], (short)0, acc[0][nt], false, false);
                acc[1][nt] = __builtin_amdgcn_wmma_f32_16x16x32_bf16(
                    false, afr[1][ks], false, bfr[nt][ks], (short)0, acc[1][nt], false, false);
            }
        }
        __builtin_amdgcn_sched_barrier(0);
    }

    // Epilogue: LeakyReLU + fp32 store
    #pragma unroll
    for (int mt = 0; mt < 2; ++mt)
        #pragma unroll
        for (int nt = 0; nt < 4; ++nt) {
            int col = cb + nt * 16 + lm;
            #pragma unroll
            for (int r = 0; r < 8; ++r) {
                int row = m0 + mt * 16 + (kh ? 8 : 0) + r;
                float v = acc[mt][nt][r];
                out[(size_t)row * C_OUT + col] = (v >= 0.f) ? v : NEG_SLOPE * v;
            }
        }
}

extern "C" void kernel_launch(void* const* d_in, const int* in_sizes, int n_in,
                              void* d_out, int out_size, void* d_ws, size_t ws_size,
                              hipStream_t stream) {
    (void)in_sizes; (void)n_in; (void)out_size; (void)ws_size;
    const float* inputs = (const float*)d_in[0];
    const float* adj    = (const float*)d_in[1];
    const float* att    = (const float*)d_in[2];
    const float* w      = (const float*)d_in[3];
    const float* bias   = (const float*)d_in[4];
    float* out = (float*)d_out;
    unsigned short* supportT = (unsigned short*)d_ws;  // 512*8192 bf16 = 8 MB

    support_kernel<<<N_NODES / 16, 256, 0, stream>>>(inputs, w, bias, supportT);
    graphconv_kernel<<<N_NODES / 32, 256, 0, stream>>>(adj, att, supportT, out);
}
